// GraphLayer_32143535243794
// MI455X (gfx1250) — compile-verified, matrix operand order checked by reading, not run on hardware
//
#include <hip/hip_runtime.h>

// ---------------------------------------------------------------------------
// GraphLayer for MI455X (gfx1250, wave32, WMMA bf16 16x16x32, f32 accum).
//
// Algebraic restructuring (exact up to bf16 rounding of GEMM operands):
//   Apre = e @ mW1[0:128]   + mb1          [N,128]
//   Bpre = e @ mW1[128:256]                [N,128]
//   h[i,j] = relu(Apre[i] + Bpre[j] + c[i,j] @ mW1[256:263])
//   hsum[i] = sum_j h[i,j]                 (reduced in registers/LDS)
//   agg  = hsum @ mW2 + N*mb2              (mW2 pulled outside the j-sum)
//   out  = relu(agg @ uW1top + e @ uW1bot + ub1) @ uW2 + ub2
// ---------------------------------------------------------------------------

typedef __attribute__((ext_vector_type(16))) __bf16 v16bf;
typedef __attribute__((ext_vector_type(8)))  float  v8f;

// Native f32 -> bf16 conversion; lowers to the hw cvt ops instead of the
// v_bfe/v_add3/v_mov_b16 manual-rounding chain seen in round 1.
__device__ __forceinline__ __bf16 f2bf(float f) { return (__bf16)f; }

// k-mapping for 16-bit WMMA A/B operands (ISA 7.12.2):
//   lanes 0-15 : VGPR v<4 -> K=2v..2v+1,  v>=4 -> K=16+2(v-4)..
//   lanes16-31 : same +8 (K=8..15, K=24..31)
__device__ __forceinline__ int klocal(int half, int t) {
    int v = t >> 1;
    return ((v >> 2) << 4) + (half << 3) + ((v & 3) << 1) + (t & 1);
}

// ---------------------------------------------------------------------------
// Pack a [rows, 128] f32 row-major weight into bf16 B-operand tiles.
// grid.x = ksteps*8 tiles; 32 threads; tile = 32 lanes x 16 bf16 (1 KB).
// ---------------------------------------------------------------------------
__global__ void pack_w(const float* __restrict__ W, int rows,
                       __bf16* __restrict__ out) {
    int tile = blockIdx.x;           // kstep*8 + chblock
    int ks   = tile >> 3;
    int b    = tile & 7;
    int lane = threadIdx.x;
    int half = lane >> 4;
    int ch   = b * 16 + (lane & 15);
    __bf16* dst = out + (size_t)tile * 512 + lane * 16;
#pragma unroll
    for (int t = 0; t < 16; ++t) {
        int k = ks * 32 + klocal(half, t);
        dst[t] = (k < rows) ? f2bf(W[(size_t)k * 128 + ch]) : f2bf(0.0f);
    }
}

// ---------------------------------------------------------------------------
// Y[i0:i0+16, :] = opt_relu( X @ Wp(bf16) + bias*bscale + Add )
// 64 blocks x 256 threads; wave w owns channel block w; 4 k-steps of 32.
// ---------------------------------------------------------------------------
__global__ void gemm128(const float* __restrict__ X,
                        const __bf16* __restrict__ Wp,
                        const float* __restrict__ bias, float bscale,
                        const float* __restrict__ add,
                        float* __restrict__ Y, int do_relu) {
    int i0   = blockIdx.x * 16;
    int w    = threadIdx.x >> 5;
    int lane = threadIdx.x & 31;
    int half = lane >> 4;
    int mrow = lane & 15;
    int ch   = w * 16 + mrow;

    v8f cc;
#pragma unroll
    for (int v = 0; v < 8; ++v) {
        float init = bias ? bias[ch] * bscale : 0.0f;
        if (add) init += add[(size_t)(i0 + v + 8 * half) * 128 + ch];
        cc[v] = init;
    }
#pragma unroll
    for (int ks = 0; ks < 4; ++ks) {
        v16bf au;
#pragma unroll
        for (int t = 0; t < 16; ++t) {
            int k = ks * 32 + klocal(half, t);
            au[t] = f2bf(X[(size_t)(i0 + mrow) * 128 + k]);
        }
        v16bf bu = *(const v16bf*)(Wp + (size_t)(ks * 8 + w) * 512 + lane * 16);
        cc = __builtin_amdgcn_wmma_f32_16x16x32_bf16(
                 false, au, false, bu, (short)0, cc, false, false);
    }
#pragma unroll
    for (int v = 0; v < 8; ++v) {
        float o = cc[v];
        if (do_relu) o = fmaxf(o, 0.0f);
        Y[(size_t)(i0 + v + 8 * half) * 128 + ch] = o;
    }
}

// ---------------------------------------------------------------------------
// Edge kernel: for each (i,j) compute h = relu(Apre[i]+Bpre[j]+c[i,j]@mW1c)
// and accumulate over j into parts[jc][i][ch].
// grid (64 i-tiles, 8 j-chunks) x 256 threads. Wave w: bh=w&1 channel half,
// js=w>>1 j-subrange (32 j each). WMMA: M=i(16), N=ch(16), K=7 padded to 32.
// Accumulator C is seeded with Apre+Bpre so the WMMA performs the bias add.
// ---------------------------------------------------------------------------
__global__ void edge_kernel(const float* __restrict__ c,
                            const float* __restrict__ Apre,
                            const float* __restrict__ Bpre,
                            const __bf16* __restrict__ WpC,
                            float* __restrict__ parts) {
    __shared__ float tile[2048];                       // 16 i x 128 ch
    int i0  = blockIdx.x * 16;
    int jc  = blockIdx.y;
    int tid = threadIdx.x;
    for (int idx = tid; idx < 2048; idx += 256) tile[idx] = 0.0f;
    __syncthreads();

    int w    = tid >> 5;
    int lane = tid & 31;
    int bh   = w & 1;                                   // channel half (64 ch)
    int js   = w >> 1;                                  // j sub-range
    int half = lane >> 4;
    int mrow = lane & 15;
    int i    = i0 + mrow;
    bool lo  = (lane < 16);

    v16bf Bw[4];
    float accA[4][8];
#pragma unroll
    for (int b2 = 0; b2 < 4; ++b2) {
        int b = bh * 4 + b2;
        Bw[b2] = *(const v16bf*)(WpC + (size_t)b * 512 + lane * 16);
        int chl = b * 16 + mrow;
#pragma unroll
        for (int v = 0; v < 8; ++v)
            accA[b2][v] = Apre[(size_t)(i0 + v + 8 * half) * 128 + chl];
    }

    float hs[4][8];
#pragma unroll
    for (int b2 = 0; b2 < 4; ++b2)
#pragma unroll
        for (int v = 0; v < 8; ++v) hs[b2][v] = 0.0f;

    int jbase = jc * 128 + js * 32;
    for (int jj = 0; jj < 32; ++jj) {
        int j = jbase + jj;
        // A operand: c[i, j, 0:7], K padded to 32 with zeros (klocal==t for t<8).
        v16bf au;
#pragma unroll
        for (int t = 0; t < 16; ++t) au[t] = f2bf(0.0f);
        if (lo) {
            const float* cr = c + ((size_t)i * 1024 + (size_t)j) * 7;
#pragma unroll
            for (int t = 0; t < 7; ++t) au[t] = f2bf(cr[t]);
        }
#pragma unroll
        for (int b2 = 0; b2 < 4; ++b2) {
            int chl  = (bh * 4 + b2) * 16 + mrow;
            float bj = Bpre[(size_t)j * 128 + chl];
            v8f cc;
#pragma unroll
            for (int v = 0; v < 8; ++v) cc[v] = accA[b2][v] + bj;
            cc = __builtin_amdgcn_wmma_f32_16x16x32_bf16(
                     false, au, false, Bw[b2], (short)0, cc, false, false);
#pragma unroll
            for (int v = 0; v < 8; ++v) hs[b2][v] += fmaxf(cc[v], 0.0f);
        }
    }

    // Reduce the 4 j-subrange waves per channel-half via LDS float atomics.
#pragma unroll
    for (int b2 = 0; b2 < 4; ++b2) {
        int chl = (bh * 4 + b2) * 16 + mrow;
#pragma unroll
        for (int v = 0; v < 8; ++v)
            atomicAdd(&tile[(v + 8 * half) * 128 + chl], hs[b2][v]);
    }
    __syncthreads();
    for (int idx = tid; idx < 2048; idx += 256)
        parts[(size_t)jc * 131072 + (size_t)(i0 + (idx >> 7)) * 128 + (idx & 127)]
            = tile[idx];
}

__global__ void reduce_parts(const float* __restrict__ parts,
                             float* __restrict__ hsum) {
    int idx = blockIdx.x * 256 + threadIdx.x;   // < 1024*128
    float s = 0.0f;
#pragma unroll
    for (int p = 0; p < 8; ++p) s += parts[(size_t)p * 131072 + idx];
    hsum[idx] = s;
}

// ---------------------------------------------------------------------------
// Workspace layout (bytes, all 1KB aligned):
//   WpA 0 (32K) | WpB 32K | WpC 64K (8K) | Wpm2 72K | Wpu1t 104K |
//   Wpu1b 136K | Wpu2 168K | buf0 200K (512K, Apre->agg) |
//   buf1 712K (512K, Bpre->tmp) | buf2 1224K (512K, hsum->t) |
//   parts 1736K (4M)   total ~5.7 MB
// ---------------------------------------------------------------------------
extern "C" void kernel_launch(void* const* d_in, const int* in_sizes, int n_in,
                              void* d_out, int out_size, void* d_ws, size_t ws_size,
                              hipStream_t stream) {
    (void)in_sizes; (void)n_in; (void)out_size; (void)ws_size;
    const float* e   = (const float*)d_in[0];
    const float* c   = (const float*)d_in[1];
    const float* mW1 = (const float*)d_in[2];
    const float* mb1 = (const float*)d_in[3];
    const float* mW2 = (const float*)d_in[4];
    const float* mb2 = (const float*)d_in[5];
    const float* uW1 = (const float*)d_in[6];
    const float* ub1 = (const float*)d_in[7];
    const float* uW2 = (const float*)d_in[8];
    const float* ub2 = (const float*)d_in[9];
    float* out = (float*)d_out;

    char* ws = (char*)d_ws;
    __bf16* WpA   = (__bf16*)(ws + 0);
    __bf16* WpB   = (__bf16*)(ws + 32768);
    __bf16* WpC   = (__bf16*)(ws + 65536);
    __bf16* Wpm2  = (__bf16*)(ws + 73728);
    __bf16* Wpu1t = (__bf16*)(ws + 106496);
    __bf16* Wpu1b = (__bf16*)(ws + 139264);
    __bf16* Wpu2  = (__bf16*)(ws + 172032);
    float* buf0  = (float*)(ws + 204800);    // Apre, later agg
    float* buf1  = (float*)(ws + 729088);    // Bpre, later tmp
    float* buf2  = (float*)(ws + 1253376);   // hsum, later t
    float* parts = (float*)(ws + 1777664);   // 8 x [1024,128]

    // 1) Pack weights into bf16 WMMA B-operand layout.
    pack_w<<<32, 32, 0, stream>>>(mW1,             128, WpA);
    pack_w<<<32, 32, 0, stream>>>(mW1 + 128 * 128, 128, WpB);
    pack_w<<< 8, 32, 0, stream>>>(mW1 + 256 * 128,   7, WpC);
    pack_w<<<32, 32, 0, stream>>>(mW2,             128, Wpm2);
    pack_w<<<32, 32, 0, stream>>>(uW1,             128, Wpu1t);
    pack_w<<<32, 32, 0, stream>>>(uW1 + 128 * 128, 128, Wpu1b);
    pack_w<<<32, 32, 0, stream>>>(uW2,             128, Wpu2);

    // 2) Apre = e @ mW1_top + mb1 ; Bpre = e @ mW1_mid
    gemm128<<<64, 256, 0, stream>>>(e, WpA, mb1, 1.0f, nullptr, buf0, 0);
    gemm128<<<64, 256, 0, stream>>>(e, WpB, nullptr, 0.0f, nullptr, buf1, 0);

    // 3) Edge sweep: hsum partials over j.
    edge_kernel<<<dim3(64, 8), 256, 0, stream>>>(c, buf0, buf1, WpC, parts);
    reduce_parts<<<512, 256, 0, stream>>>(parts, buf2);

    // 4) agg = hsum @ mW2 + N*mb2
    gemm128<<<64, 256, 0, stream>>>(buf2, Wpm2, mb2, 1024.0f, nullptr, buf0, 0);

    // 5) t = relu(agg @ uW1_top + e @ uW1_bot + ub1) ; out = t @ uW2 + ub2
    gemm128<<<64, 256, 0, stream>>>(buf0, Wpu1t, nullptr, 0.0f, nullptr, buf1, 0);
    gemm128<<<64, 256, 0, stream>>>(e,    Wpu1b, ub1, 1.0f, buf1, buf2, 1);
    gemm128<<<64, 256, 0, stream>>>(buf2, Wpu2,  ub2, 1.0f, nullptr, out, 0);
}
